// UncGuidedWindowFFTEenhance3D_21861383536947
// MI455X (gfx1250) — matrix operand order repaired and checked by reading, not compiled, not run on hardware
//
#include <hip/hip_runtime.h>
#include <hip/hip_bf16.h>
#include <math.h>
#include <stdint.h>

// ---------------------------------------------------------------------------
// UncGuidedWindowFFTEnhance3D for gfx1250 (MI455X)
//   freq-mask enhance == circular conv with g_n = tanh(gamma)*s_n*sum_k w_nk g_k
//   -> per-window 216x216 circulant GEMM done with v_wmma_f32_16x16x32_bf16
//   Circulant index pattern is window-invariant -> precomputed uint8 LUT.
// ---------------------------------------------------------------------------

typedef __attribute__((ext_vector_type(16))) __bf16 v16bf;
typedef __attribute__((ext_vector_type(8)))  float  v8f;

#define CCH   32
#define NSIDE 96
#define HWSZ  (96 * 96)           // 9216
#define DHW   (96 * 96 * 96)      // 884736
#define NWIN  4096                // 16^3 windows
#define VOL   216                 // 6^3
#define KBAS  8
#define PADK  224                 // 216 padded to 14*16

__device__ __forceinline__ v8f wmma_bf16f32(v16bf a, v16bf b, v8f c) {
  // D(f32 16x16) = A(bf16 16x32) * B(bf16 32x16) + C
  return __builtin_amdgcn_wmma_f32_16x16x32_bf16(false, a, false, b, (short)0, c,
                                                 false, false);
}

// A-fragment (16x32 bf16) from row-major [rows][ld], k offset kk.
// lane<16: row=lane, K = kk+{0..15}; lane>=16: row=lane-16,
// K = kk+{16..31}. One contiguous 32B read per lane.
__device__ __forceinline__ v16bf frag_a(const __bf16* A, int ld, int lane, int kk) {
  int r  = lane & 15;
  int ko = kk + ((lane & 16) ? 16 : 0);
  const __bf16* p = A + r * ld + ko;
  v16bf a;
#pragma unroll
  for (int i = 0; i < 16; ++i) a[i] = p[i];
  return a;
}

// B-fragment (32x16 bf16) from a TRANSPOSED (N-major) [N][ldk] array:
// lane<16: N=n0+lane, K=kk+{0..15}; lane>=16: N=n0+lane-16, K=kk+{16..31}.
// One contiguous 32B read per lane.
__device__ __forceinline__ v16bf frag_bT(const __bf16* Bt, int ldk, int lane,
                                         int kk, int n0) {
  int col = n0 + (lane & 15);
  int kb  = kk + ((lane & 16) ? 16 : 0);
  const __bf16* p = Bt + col * ldk + kb;
  v16bf b;
#pragma unroll
  for (int i = 0; i < 16; ++i) b[i] = p[i];
  return b;
}

__device__ __forceinline__ float gelu_exact(float x) {
  return 0.5f * x * (1.0f + erff(x * 0.70710678118654752f));
}

__device__ __forceinline__ uint32_t bf16bits(float x) {
  return (uint32_t)__builtin_bit_cast(unsigned short, (__bf16)x);
}

// ---------------------------------------------------------------------------
// Kernel 1: prep — conv kernels g[8][216], tanh(gamma), bf16 transposed
//           weights, and the window-invariant circulant gather LUT.
// ---------------------------------------------------------------------------
__global__ __launch_bounds__(256) void k_prep(const float* __restrict__ mu_raw,
                                              const float* __restrict__ sigma_raw,
                                              const float* __restrict__ gamma,
                                              const float* __restrict__ W1,
                                              const float* __restrict__ W2,
                                              const float* __restrict__ Wb,
                                              const float* __restrict__ Ws,
                                              float* __restrict__ g,
                                              float* __restrict__ tg,
                                              __bf16* __restrict__ W1t,
                                              __bf16* __restrict__ W2t,
                                              __bf16* __restrict__ W3t,
                                              uint8_t* __restrict__ idx8) {
  __shared__ float basisS[KBAS][144];   // [k][d*24 + h*4 + w] on rfft grid 6x6x4
  __shared__ float maskS[KBAS][VOL];    // mirrored to full 6x6x6 grid
  const int t = threadIdx.x;
  if (t == 0) tg[0] = tanhf(gamma[0]);

  // ---- weight transpose + bf16 convert ----
  for (int i = t; i < 64 * 128; i += 256) {      // W1t[col][k], K padded 97->128
    int col = i >> 7, k = i & 127;
    W1t[col * 128 + k] = (k < 97) ? (__bf16)W1[k * 64 + col] : (__bf16)0.0f;
  }
  for (int i = t; i < 64 * 64; i += 256) {       // W2t[col][k]
    int col = i >> 6, k = i & 63;
    W2t[col * 64 + k] = (__bf16)W2[k * 64 + col];
  }
  for (int i = t; i < 16 * 64; i += 256) {       // W3t[col][k]: 0..7 Wb, 8 Ws
    int col = i >> 6, k = i & 63;
    float v = (col < 8) ? Wb[k * 8 + col] : ((col == 8) ? Ws[k] : 0.0f);
    W3t[col * 64 + k] = (__bf16)v;
  }

  // ---- circulant gather LUT: A[x][s] = kern[idx8[x*224+s]], 216 == zero ----
  for (int i = t; i < PADK * PADK; i += 256) {
    int x = i / PADK, s = i % PADK;
    uint8_t v = (uint8_t)216;  // sentinel -> 0.0
    if (x < VOL && s < VOL) {
      int xz = x / 36, xy = (x / 6) % 6, xx = x % 6;
      int sz = s / 36, sy = (s / 6) % 6, sx = s % 6;
      int dz = xz - sz; dz += (dz < 0) ? 6 : 0;
      int dy = xy - sy; dy += (dy < 0) ? 6 : 0;
      int dx = xx - sx; dx += (dx < 0) ? 6 : 0;
      v = (uint8_t)(dz * 36 + dy * 6 + dx);
    }
    idx8[i] = v;
  }

  // ---- basis on rfft grid ----
  const float FA[6] = {0.0f, 1.0f / 6.0f, 1.0f / 3.0f, 0.5f, 1.0f / 3.0f, 1.0f / 6.0f};
  const float FR[4] = {0.0f, 1.0f / 6.0f, 1.0f / 3.0f, 0.5f};

  if (t < 144) {
    int d = t / 24, h = (t / 4) % 6, w = t % 4;
    float fd = FA[d], fh = FA[h], fw = FR[w];
    float rr = sqrtf(fd * fd + fh * fh + fw * fw) / (0.8660254037844386f + 1e-8f);
    float bv[KBAS];
    float bsum = 0.0f;
#pragma unroll
    for (int k = 0; k < KBAS; ++k) {
      float mu  = 1.0f / (1.0f + expf(-mu_raw[k]));
      float sig = log1pf(expf(sigma_raw[k])) + 0.02f;   // softplus + MIN_SIG
      float zz  = (rr - mu) / (sig + 1e-8f);
      bv[k] = expf(-0.5f * zz * zz);
      bsum += bv[k];
    }
    bsum = fmaxf(bsum, 1e-8f);
#pragma unroll
    for (int k = 0; k < KBAS; ++k) basisS[k][t] = bv[k] / bsum;
  }
  __syncthreads();

  for (int i = t; i < KBAS * VOL; i += 256) {
    int k = i / VOL, r = i % VOL;
    int d = r / 36, h = (r / 6) % 6, w = r % 6;
    int wf = (w <= 3) ? w : 6 - w;  // conjugate-symmetric mirror on last axis
    maskS[k][r] = basisS[k][d * 24 + h * 4 + wf];
  }
  __syncthreads();

  // g_k[z,y,x] = (1/216) sum_f mask_k(f) cos(2*pi*(f.d)/6)   (mask is symmetric)
  const float C6[6] = {1.0f, 0.5f, -0.5f, -1.0f, -0.5f, 0.5f};
  for (int i = t; i < KBAS * VOL; i += 256) {
    int k = i / VOL, r = i % VOL;
    int z = r / 36, y = (r / 6) % 6, x = r % 6;
    float acc = 0.0f;
    for (int f0 = 0; f0 < 6; ++f0)
      for (int f1 = 0; f1 < 6; ++f1) {
        int m01 = f0 * z + f1 * y;
        const float* mrow = &maskS[k][f0 * 36 + f1 * 6];
#pragma unroll
        for (int f2 = 0; f2 < 6; ++f2)
          acc += mrow[f2] * C6[(m01 + f2 * x) % 6];
      }
    g[k * VOL + r] = acc * (1.0f / 216.0f);
  }
}

// ---------------------------------------------------------------------------
// Kernel 2: per-window stats -> cond[N][128] (bf16, zero-padded K)
// ---------------------------------------------------------------------------
__global__ __launch_bounds__(256) void k_stats(const float* __restrict__ low,
                                               const float* __restrict__ gate,
                                               const float* __restrict__ unc,
                                               __bf16* __restrict__ cond) {
  __shared__ float sG[CCH][8], sM[CCH][8], sV[CCH][8], sU[8];
  const int n = blockIdx.x;
  const int bz = n / 256, by = (n / 16) % 16, bx = n % 16;
  const size_t bOff = (size_t)(bz * 6) * HWSZ + (size_t)(by * 6) * NSIDE + (size_t)(bx * 6);
  const int t = threadIdx.x;
  const int c = t >> 3, s = t & 7;

  float ag = 0.0f, am = 0.0f, av = 0.0f, au = 0.0f;
  for (int j = s * 27; j < s * 27 + 27; ++j) {
    int dz = j / 36, dy = (j / 6) % 6, dx = j % 6;
    size_t sp = bOff + (size_t)dz * HWSZ + (size_t)dy * NSIDE + dx;
    float lv = low[(size_t)c * DHW + sp];
    ag += gate[(size_t)c * DHW + sp];
    am += lv;
    av += lv * lv;
    if (c == 0) au += fminf(fmaxf(unc[sp], 0.0f), 1.0f);
  }
  sG[c][s] = ag; sM[c][s] = am; sV[c][s] = av;
  if (c == 0) sU[s] = au;
  __syncthreads();

  __bf16* row = cond + (size_t)n * 128;
  if (t < CCH) {
    float gs = 0.0f, ms = 0.0f, vs = 0.0f;
#pragma unroll
    for (int i = 0; i < 8; ++i) { gs += sG[t][i]; ms += sM[t][i]; vs += sV[t][i]; }
    float mean = ms * (1.0f / 216.0f);
    float var  = vs * (1.0f / 216.0f) - mean * mean;
    row[t]      = (__bf16)(gs * (1.0f / 216.0f));
    row[32 + t] = (__bf16)mean;
    row[64 + t] = (__bf16)sqrtf(fmaxf(var, 1e-8f));
  }
  if (t == 0) {
    float us = 0.0f;
#pragma unroll
    for (int i = 0; i < 8; ++i) us += sU[i];
    row[96] = (__bf16)(us * (1.0f / 216.0f));
  }
  if (t >= 97 && t < 128) row[t] = (__bf16)0.0f;   // K padding
}

// ---------------------------------------------------------------------------
// Kernel 3: fused MLP (WMMA bf16): 16 windows per wave/block
// ---------------------------------------------------------------------------
__global__ __launch_bounds__(32) void k_mlp(const __bf16* __restrict__ cond,
                                            const __bf16* __restrict__ W1t,
                                            const float* __restrict__ b1,
                                            const __bf16* __restrict__ W2t,
                                            const float* __restrict__ b2,
                                            const __bf16* __restrict__ W3t,
                                            const float* __restrict__ bb,
                                            const float* __restrict__ bs,
                                            float* __restrict__ wmix,
                                            float* __restrict__ strength) {
  __shared__ __bf16 hS[16][64];
  __shared__ float  lg[16][9];
  const int lane = threadIdx.x;
  const int m0   = blockIdx.x * 16;
  const __bf16* Arows = cond + (size_t)m0 * 128;

  const int cn = lane & 15;
  const int rb = (lane & 16) ? 8 : 0;

  // ---- layer 1: [16x128] x [128x64] ----
  v8f acc1[4] = {{0,0,0,0,0,0,0,0},{0,0,0,0,0,0,0,0},{0,0,0,0,0,0,0,0},{0,0,0,0,0,0,0,0}};
#pragma unroll
  for (int kk = 0; kk < 128; kk += 32) {
    v16bf a = frag_a(Arows, 128, lane, kk);
#pragma unroll
    for (int nt = 0; nt < 4; ++nt)
      acc1[nt] = wmma_bf16f32(a, frag_bT(W1t, 128, lane, kk, nt * 16), acc1[nt]);
  }
#pragma unroll
  for (int nt = 0; nt < 4; ++nt)
#pragma unroll
    for (int j = 0; j < 8; ++j) {
      float x = acc1[nt][j] + b1[nt * 16 + cn];
      hS[rb + j][nt * 16 + cn] = (__bf16)gelu_exact(x);
    }
  __syncthreads();

  // ---- layer 2: [16x64] x [64x64] ----
  v8f acc2[4] = {{0,0,0,0,0,0,0,0},{0,0,0,0,0,0,0,0},{0,0,0,0,0,0,0,0},{0,0,0,0,0,0,0,0}};
#pragma unroll
  for (int kk = 0; kk < 64; kk += 32) {
    v16bf a = frag_a(&hS[0][0], 64, lane, kk);
#pragma unroll
    for (int nt = 0; nt < 4; ++nt)
      acc2[nt] = wmma_bf16f32(a, frag_bT(W2t, 64, lane, kk, nt * 16), acc2[nt]);
  }
  __syncthreads();
#pragma unroll
  for (int nt = 0; nt < 4; ++nt)
#pragma unroll
    for (int j = 0; j < 8; ++j) {
      float x = acc2[nt][j] + b2[nt * 16 + cn];
      hS[rb + j][nt * 16 + cn] = (__bf16)gelu_exact(x);
    }
  __syncthreads();

  // ---- layer 3: [16x64] x [64x16]  (cols 0..7 = Wb, col 8 = Ws) ----
  v8f acc3 = {0,0,0,0,0,0,0,0};
#pragma unroll
  for (int kk = 0; kk < 64; kk += 32) {
    v16bf a = frag_a(&hS[0][0], 64, lane, kk);
    acc3 = wmma_bf16f32(a, frag_bT(W3t, 64, lane, kk, 0), acc3);
  }
  if (cn < 9) {
#pragma unroll
    for (int j = 0; j < 8; ++j)
      lg[rb + j][cn] = acc3[j] + ((cn < 8) ? bb[cn] : bs[0]);
  }
  __syncthreads();

  if (lane < 16) {
    const int r = m0 + lane;
    float mx = -3.4e38f;
#pragma unroll
    for (int k = 0; k < 8; ++k) mx = fmaxf(mx, lg[lane][k]);
    float e[8], se = 0.0f;
#pragma unroll
    for (int k = 0; k < 8; ++k) { e[k] = expf(lg[lane][k] - mx); se += e[k]; }
    float inv = 1.0f / se;
#pragma unroll
    for (int k = 0; k < 8; ++k) wmix[r * 8 + k] = e[k] * inv;
    strength[r] = 1.0f / (1.0f + expf(-lg[lane][8]));
  }
}

// ---------------------------------------------------------------------------
// Kernel 4: per-window circulant GEMM apply (WMMA bf16), residual add
// ---------------------------------------------------------------------------
__global__ __launch_bounds__(256) void k_apply(const float* __restrict__ low,
                                               const float* __restrict__ g,
                                               const float* __restrict__ tg,
                                               const float* __restrict__ wmix,
                                               const float* __restrict__ strength,
                                               const uint8_t* __restrict__ idx8,
                                               float* __restrict__ out) {
  __shared__ __bf16   Amat[PADK * PADK];   // 224x224 circulant (bf16)    ~98KB
  __shared__ __bf16   XmatT[CCH * PADK];   // 32x224  (N-major, contig K) ~14KB
  __shared__ float    Lw[CCH][VOL];        // f32 copy for residual       ~27KB
  __shared__ uint32_t gsbU[256];           // mixed kernel, raw bf16 bits

  const int n = blockIdx.x;
  const int bz = n / 256, by = (n / 16) % 16, bx = n % 16;
  const size_t bOff = (size_t)(bz * 6) * HWSZ + (size_t)(by * 6) * NSIDE + (size_t)(bx * 6);
  const int t = threadIdx.x;

  // mix per-window kernel, fold tanh(gamma)*strength, keep raw bf16 bits
  if (t < VOL) {
    float acc = 0.0f;
#pragma unroll
    for (int k = 0; k < KBAS; ++k) acc += wmix[n * 8 + k] * g[k * VOL + t];
    gsbU[t] = bf16bits(acc * tg[0] * strength[n]);
  } else {
    gsbU[t] = 0u;  // sentinel region (216..255) -> +0.0 bf16
  }
  for (int i = t; i < CCH * VOL; i += 256) {
    int c = i / VOL, s = i % VOL;
    int dz = s / 36, dy = (s / 6) % 6, dx = s % 6;
    float v = low[(size_t)c * DHW + bOff + (size_t)dz * HWSZ + (size_t)dy * NSIDE + dx];
    Lw[c][s] = v;
    XmatT[c * PADK + s] = (__bf16)v;
  }
  // zero K-pad cols 216..223 of X^T
  {
    int c = t >> 3, k = VOL + (t & 7);
    XmatT[c * PADK + k] = (__bf16)0.0f;
  }
  __syncthreads();

  // circulant fill via precomputed LUT: 16 bytes of indices -> 16 bf16 values
  {
    const uint32_t* lutw = (const uint32_t*)idx8;
    uint32_t* Aw = (uint32_t*)&Amat[0];
    const int NCHUNK = (PADK * PADK) / 16;  // 3136
    for (int ch = t; ch < NCHUNK; ch += 256) {
      const uint32_t* p = lutw + ch * 4;
      uint32_t w0 = p[0], w1 = p[1], w2 = p[2], w3 = p[3];
      uint32_t* dst = Aw + ch * 8;
      uint32_t wq[4] = {w0, w1, w2, w3};
#pragma unroll
      for (int q = 0; q < 4; ++q) {
        uint32_t w = wq[q];
        uint32_t e0 = gsbU[w & 255u];
        uint32_t e1 = gsbU[(w >> 8) & 255u];
        uint32_t e2 = gsbU[(w >> 16) & 255u];
        uint32_t e3 = gsbU[w >> 24];
        dst[q * 2 + 0] = e0 | (e1 << 16);
        dst[q * 2 + 1] = e2 | (e3 << 16);
      }
    }
  }
  __syncthreads();

  // D(224x32) = A(224x224) * X(224x32): 14x2 tiles over 8 waves, 7 K-steps
  const int wave = t >> 5, lane = t & 31;
  for (int tile = wave; tile < 28; tile += 8) {
    const int mt = tile >> 1, nt = tile & 1;
    v8f acc = {0, 0, 0, 0, 0, 0, 0, 0};
#pragma unroll
    for (int kk = 0; kk < PADK; kk += 32) {
      v16bf a = frag_a(&Amat[mt * 16 * PADK], PADK, lane, kk);
      v16bf b = frag_bT(&XmatT[0], PADK, lane, kk, nt * 16);
      acc = wmma_bf16f32(a, b, acc);
    }
    const int c  = nt * 16 + (lane & 15);
    const int rbase = mt * 16 + ((lane & 16) ? 8 : 0);
#pragma unroll
    for (int j = 0; j < 8; ++j) {
      int m = rbase + j;
      if (m < VOL) {
        int dz = m / 36, dy = (m / 6) % 6, dx = m % 6;
        out[(size_t)c * DHW + bOff + (size_t)dz * HWSZ + (size_t)dy * NSIDE + dx] =
            Lw[c][m] + acc[j];
      }
    }
  }
}

// ---------------------------------------------------------------------------
extern "C" void kernel_launch(void* const* d_in, const int* in_sizes, int n_in,
                              void* d_out, int out_size, void* d_ws, size_t ws_size,
                              hipStream_t stream) {
  const float* low       = (const float*)d_in[0];
  const float* gate_dec  = (const float*)d_in[1];
  const float* unc       = (const float*)d_in[2];
  const float* W1        = (const float*)d_in[3];
  const float* b1        = (const float*)d_in[4];
  const float* W2        = (const float*)d_in[5];
  const float* b2        = (const float*)d_in[6];
  const float* Wb        = (const float*)d_in[7];
  const float* bb        = (const float*)d_in[8];
  const float* Ws        = (const float*)d_in[9];
  const float* bs        = (const float*)d_in[10];
  const float* mu_raw    = (const float*)d_in[11];
  const float* sigma_raw = (const float*)d_in[12];
  const float* gamma     = (const float*)d_in[13];
  float* out = (float*)d_out;

  char* ws = (char*)d_ws;
  float*   g_ws    = (float*)(ws + 0);                      // 8*216 f32
  float*   tg_ws   = (float*)(ws + 8192);                   // 1 f32
  float*   wmix_ws = (float*)(ws + 8448);                   // 4096*8 f32
  float*   str_ws  = (float*)(ws + 8448 + 131072);          // 4096 f32
  __bf16*  cond_ws = (__bf16*)(ws + 155904);                // 4096*128 bf16
  __bf16*  W1t_ws  = (__bf16*)(ws + 1204480);               // 64*128 bf16
  __bf16*  W2t_ws  = (__bf16*)(ws + 1220864);               // 64*64  bf16
  __bf16*  W3t_ws  = (__bf16*)(ws + 1229056);               // 16*64  bf16
  uint8_t* idx_ws  = (uint8_t*)(ws + 1231104);              // 224*224 u8

  k_prep<<<1, 256, 0, stream>>>(mu_raw, sigma_raw, gamma, W1, W2, Wb, Ws,
                                g_ws, tg_ws, W1t_ws, W2t_ws, W3t_ws, idx_ws);
  k_stats<<<NWIN, 256, 0, stream>>>(low, gate_dec, unc, cond_ws);
  k_mlp<<<NWIN / 16, 32, 0, stream>>>(cond_ws, W1t_ws, b1, W2t_ws, b2, W3t_ws,
                                      bb, bs, wmix_ws, str_ws);
  k_apply<<<NWIN, 256, 0, stream>>>(low, g_ws, tg_ws, wmix_ws, str_ws, idx_ws, out);
}
